// SlidingWindowAttention_32762010534333
// MI455X (gfx1250) — matrix operand order repaired and checked by reading, not compiled, not run on hardware
//
#include <hip/hip_runtime.h>
#include <math.h>

// ---------------- problem constants (reference: H=16, HKV=4, D=128, W=1024) --
constexpr int T_LEN = 4096;
constexpr int NEMB  = 2048;
constexpr int NH    = 16;
constexpr int NKV   = 4;
constexpr int HD    = 128;
constexpr int WIN   = 1024;

constexpr int AS = 48;  // GEMM LDS A row stride (halves): 96B, 16B-aligned
constexpr int BS = 48;  // GEMM LDS B row stride (halves)

typedef __attribute__((ext_vector_type(8)))  _Float16     v8h;
typedef __attribute__((ext_vector_type(16))) _Float16     v16h;
typedef __attribute__((ext_vector_type(8)))  float        v8f;
typedef __attribute__((ext_vector_type(4)))  unsigned int u32x4;
typedef __attribute__((ext_vector_type(8)))  unsigned int u32x8;

__device__ __forceinline__ v16h cat8(v8h lo, v8h hi) {
  return __builtin_shufflevector(lo, hi, 0,1,2,3,4,5,6,7,8,9,10,11,12,13,14,15);
}
__device__ __forceinline__ v8f wmma_f16(v16h a, v16h b, v8f c) {
  // emits v_wmma_f32_16x16x32_f16 (wave32, f32 accumulate)
  return __builtin_amdgcn_wmma_f32_16x16x32_f16(false, a, false, b, (short)0, c,
                                                false, false);
}
__device__ __forceinline__ v8f zero8f() {
  v8f r;
#pragma unroll
  for (int i = 0; i < 8; ++i) r[i] = 0.0f;
  return r;
}
__device__ __forceinline__ v8h cvt2(float4 a, float4 b) {
  v8h r;
  r[0] = (_Float16)a.x; r[1] = (_Float16)a.y; r[2] = (_Float16)a.z; r[3] = (_Float16)a.w;
  r[4] = (_Float16)b.x; r[5] = (_Float16)b.y; r[6] = (_Float16)b.z; r[7] = (_Float16)b.w;
  return r;
}

// ---- A-tile register staging, overloaded on element type -------------------
template <typename AT> struct ARegsT { float4 r[4]; };
template <>            struct ARegsT<_Float16> { v8h r[2]; };

__device__ __forceinline__ void fetchA(const float* p, float4 (&r)[4]) {
  const float4* a4 = (const float4*)p;
  r[0] = a4[0]; r[1] = a4[1]; r[2] = a4[2]; r[3] = a4[3];
}
__device__ __forceinline__ void fetchA(const _Float16* p, v8h (&r)[2]) {
  const v8h* a8 = (const v8h*)p;
  r[0] = a8[0]; r[1] = a8[1];
}
__device__ __forceinline__ void storeA(_Float16* d, const float4 (&r)[4]) {
  *(v8h*)d = cvt2(r[0], r[1]); *(v8h*)(d + 8) = cvt2(r[2], r[3]);
}
__device__ __forceinline__ void storeA(_Float16* d, const v8h (&r)[2]) {
  *(v8h*)d = r[0]; *(v8h*)(d + 8) = r[1];
}

// ======================================================================
// Tensor Data Mover: 2D tile load (global f16 -> LDS), 2-SGPR-group D#
// (VADDR2/VADDR3 = NULL form, valid for tensors up to 2D).
// Reads past tensor_dim return zero. Tracked by TENSORcnt (in-order per
// wave), so double-buffering is a simple s_wait_tensorcnt(2).
// ======================================================================
__device__ __forceinline__ void tdm_load_2d(
    unsigned lds_byte_addr, unsigned long long gaddr,
    unsigned tensor_d0, unsigned tensor_d1, unsigned d0_stride,
    unsigned tile_d0, unsigned tile_d1) {
  u32x4 g0;
  g0[0] = 1u;                                   // count=1, user descriptor
  g0[1] = lds_byte_addr;                        // D#.lds_addr  (bits 63:32)
  g0[2] = (unsigned)(gaddr & 0xffffffffu);      // global_addr[31:0]
  g0[3] = ((unsigned)(gaddr >> 32) & 0x01ffffffu) | (2u << 30);  // addr[56:32] | type=2
  u32x8 g1;
  g1[0] = 1u << 16;                             // data_size = 1 -> 2 bytes/elem
  g1[1] = (tensor_d0 & 0xffffu) << 16;          // tensor_dim0[15:0] @ bits 63:48
  g1[2] = (tensor_d0 >> 16) | ((tensor_d1 & 0xffffu) << 16);
  g1[3] = (tensor_d1 >> 16) | (tile_d0 << 16);  // tile_dim0 @ bits 127:112
  g1[4] = tile_d1;                              // tile_dim1; tile_dim2 = 0
  g1[5] = d0_stride;                            // tensor_dim0_stride[31:0]
  g1[6] = 0u;
  g1[7] = 0u;
  asm volatile("tensor_load_to_lds %0, %1" :: "s"(g0), "s"(g1) : "memory");
}

// Issue K tile (32 keys x 128 d -> [key][d]) and V tile (128 d x 32 keys
// from pre-transposed (HKV,D,T) global -> [d][key]) for one key tile.
__device__ __forceinline__ void attn_issue_tiles(
    const _Float16* kh, const _Float16* vtg, int kvh, int keyBase,
    unsigned ldsK, unsigned ldsV) {
  tdm_load_2d(ldsK,
              (unsigned long long)(const void*)(kh + ((size_t)keyBase * NKV + kvh) * HD),
              /*tensor_d0=*/HD, /*tensor_d1=*/(unsigned)(T_LEN - keyBase),
              /*d0_stride=*/NKV * HD, /*tile_d0=*/HD, /*tile_d1=*/32);
  tdm_load_2d(ldsV,
              (unsigned long long)(const void*)(vtg + (size_t)kvh * HD * T_LEN + keyBase),
              /*tensor_d0=*/(unsigned)(T_LEN - keyBase), /*tensor_d1=*/HD,
              /*d0_stride=*/T_LEN, /*tile_d0=*/32, /*tile_d1=*/HD);
}

// ======================================================================
// WMMA GEMM:  C[M,N] = A[M,K] * B[K,N]   (A fp32 or f16; B fp32; C fp32)
// Block tile 256x128, K-step 32, 256 threads = 8 waves, wave tile 64x64
// (4x4 WMMA accumulators; B fragments register-reused across 4 WMMAs).
// Double-buffered LDS, one barrier per K-step; tile k+1's global loads
// are issued before the WMMAs of tile k to hide HBM latency.
// Requires M % 256 == 0, N % 128 == 0, K % 32 == 0.
// ======================================================================
template <int N, int K, typename AT>
__global__ __launch_bounds__(256) void gemm_wmma(
    const AT* __restrict__ A, const float* __restrict__ B,
    float* __restrict__ C) {
  constexpr int NT = K / 32;
  __shared__ _Float16 As[2][256 * AS];  // [m][k]
  __shared__ _Float16 Bs[2][128 * BS];  // [n][k]  (transposed)

  const int tid  = threadIdx.x;
  const int lane = tid & 31;
  const int wv   = tid >> 5;
  const int l15  = lane & 15;
  const int sub  = lane >> 4;

  const int bm = blockIdx.y * 256;
  const int bn = blockIdx.x * 128;
  const int wm = (wv & 3) * 64;   // wave row offset within tile
  const int wn = (wv >> 2) * 64;  // wave col offset within tile

  // staging thread mapping: A = two passes of 128 half-rows; B = 16 cols/k-row
  const int arow = tid >> 1, ac0 = (tid & 1) * 16;
  const int bk   = tid >> 3, bn0 = (tid & 7) * 16;
  const AT* aptr = A + (size_t)(bm + arow) * K + ac0;
  const size_t aPass = (size_t)128 * K;  // +128 rows
  const float* bptr = B + (size_t)bk * N + bn + bn0;

  v8f acc[4][4];
#pragma unroll
  for (int mt = 0; mt < 4; ++mt)
#pragma unroll
    for (int nt = 0; nt < 4; ++nt) acc[mt][nt] = zero8f();

  // ---- prologue: stage tile 0 into buffer 0 ----
  {
    ARegsT<AT> t0, t1;
    fetchA(aptr, t0.r);
    fetchA(aptr + aPass, t1.r);
    const float4* b4 = (const float4*)bptr;
    float4 rb0 = b4[0], rb1 = b4[1], rb2 = b4[2], rb3 = b4[3];
    storeA(&As[0][arow * AS + ac0], t0.r);
    storeA(&As[0][(arow + 128) * AS + ac0], t1.r);
    float tmp[16] = {rb0.x, rb0.y, rb0.z, rb0.w, rb1.x, rb1.y, rb1.z, rb1.w,
                     rb2.x, rb2.y, rb2.z, rb2.w, rb3.x, rb3.y, rb3.z, rb3.w};
#pragma unroll
    for (int j = 0; j < 16; ++j) Bs[0][(bn0 + j) * BS + bk] = (_Float16)tmp[j];
  }
  __syncthreads();

  for (int it = 0; it < NT; ++it) {
    const int cur = it & 1;
    const bool more = (it + 1 < NT);

    // ---- issue global loads for tile it+1 (latency overlapped w/ WMMA) ----
    ARegsT<AT> t0, t1;
    float4 rb0, rb1, rb2, rb3;
    if (more) {
      fetchA(aptr + (it + 1) * 32, t0.r);
      fetchA(aptr + (it + 1) * 32 + aPass, t1.r);
      const float4* b4 = (const float4*)(bptr + (size_t)(it + 1) * 32 * N);
      rb0 = b4[0]; rb1 = b4[1]; rb2 = b4[2]; rb3 = b4[3];
    }

    // ---- 16 WMMAs from buffer `cur` ----
    v16h af[4];
#pragma unroll
    for (int mt = 0; mt < 4; ++mt) {
      const _Float16* p = &As[cur][(wm + mt * 16 + l15) * AS + sub * 8];
      af[mt] = cat8(*(const v8h*)p, *(const v8h*)(p + 16));
    }
#pragma unroll
    for (int nt = 0; nt < 4; ++nt) {
      const _Float16* p = &Bs[cur][(wn + nt * 16 + l15) * BS + sub * 8];
      v16h bf = cat8(*(const v8h*)p, *(const v8h*)(p + 16));
#pragma unroll
      for (int mt = 0; mt < 4; ++mt) acc[mt][nt] = wmma_f16(af[mt], bf, acc[mt][nt]);
    }

    // ---- store prefetched tile into the other buffer ----
    if (more) {
      storeA(&As[cur ^ 1][arow * AS + ac0], t0.r);
      storeA(&As[cur ^ 1][(arow + 128) * AS + ac0], t1.r);
      float tmp[16] = {rb0.x, rb0.y, rb0.z, rb0.w, rb1.x, rb1.y, rb1.z, rb1.w,
                       rb2.x, rb2.y, rb2.z, rb2.w, rb3.x, rb3.y, rb3.z, rb3.w};
#pragma unroll
      for (int j = 0; j < 16; ++j)
        Bs[cur ^ 1][(bn0 + j) * BS + bk] = (_Float16)tmp[j];
    }
    __syncthreads();
  }

  // ---- epilogue: C layout lane=N col, rows sub*8+i; i*N folds to immediates --
#pragma unroll
  for (int mt = 0; mt < 4; ++mt)
#pragma unroll
    for (int nt = 0; nt < 4; ++nt) {
      const int n     = bn + wn + nt * 16 + l15;
      const int rbase = bm + wm + mt * 16 + sub * 8;
      float* cp = C + (size_t)rbase * N + n;
#pragma unroll
      for (int i = 0; i < 8; ++i) cp[i * N] = acc[mt][nt][i];
    }
}

// ======================================================================
// RoPE + RMSNorm + quant-dequant.  One block (128 threads) per (t, head).
// mode 0: rope+rmsnorm (q)          -> out[(t,h,d)]
// mode 1: rope+rmsnorm+qdq (k)      -> out[(t,h,d)]
// mode 2: qdq only (v)              -> out[(h,d,t)]  (transposed for TDM)
// ======================================================================
__global__ __launch_bounds__(128) void rope_norm_quant(
    const float* __restrict__ in, const float* __restrict__ cosb,
    const float* __restrict__ sinb, _Float16* __restrict__ outp,
    int heads, int mode) {
  const int t = blockIdx.x, hh = blockIdx.y, d = threadIdx.x;
  const int lane = d & 31, wid = d >> 5;
  __shared__ float buf[HD];
  __shared__ float redA[4];
  __shared__ float redB[4];

  const size_t idx = ((size_t)t * heads + hh) * HD + d;
  float x = in[idx];

  if (mode != 2) {
    buf[d] = x;
    __syncthreads();
    const int j = d & 63;
    const float c = cosb[t * 64 + j];
    const float s = sinb[t * 64 + j];
    if (d < 64) x =  buf[d] * c + buf[d + 64] * s;
    else        x = -buf[d - 64] * s + buf[d] * c;
    // rmsnorm over 128 elements
    float v = x * x;
    v += __shfl_xor(v, 1, 32);  v += __shfl_xor(v, 2, 32);
    v += __shfl_xor(v, 4, 32);  v += __shfl_xor(v, 8, 32);
    v += __shfl_xor(v, 16, 32);
    if (lane == 0) redA[wid] = v;
    __syncthreads();
    const float tot = redA[0] + redA[1] + redA[2] + redA[3];
    x *= rsqrtf(tot * (1.0f / 128.0f) + 1.1920929e-07f);
  }
  if (mode != 0) {
    float a = fabsf(x);
    a = fmaxf(a, __shfl_xor(a, 1, 32));  a = fmaxf(a, __shfl_xor(a, 2, 32));
    a = fmaxf(a, __shfl_xor(a, 4, 32));  a = fmaxf(a, __shfl_xor(a, 8, 32));
    a = fmaxf(a, __shfl_xor(a, 16, 32));
    if (lane == 0) redB[wid] = a;
    __syncthreads();
    const float am = fmaxf(fmaxf(redB[0], redB[1]), fmaxf(redB[2], redB[3]));
    const float sc = fmaxf(am * (1.0f / 3.0f), 1e-8f);
    const float code = fminf(fmaxf(rintf(x / sc), -3.0f), 3.0f);
    x = code * sc;
  }
  if (mode == 2)
    outp[((size_t)hh * HD + d) * T_LEN + t] = (_Float16)x;  // (HKV, D, T)
  else
    outp[idx] = (_Float16)x;
}

// ======================================================================
// Sliding-window flash attention. Block = 128 queries (8 waves x 16),
// one (queryTile, head) per block. 32-key tiles over [qBase-W+1, qBase+127].
// K/V tiles staged by the Tensor Data Mover, double-buffered: wave 0 issues
// tile i+1's DMAs then s_wait_tensorcnt(2) so only tile i is guaranteed
// complete (TENSORcnt is in-order). pT is wave-private -> no mid barrier.
// Output written as f16 (feeds the f16-A Wo GEMM directly).
// ======================================================================
__global__ __launch_bounds__(256) void swa_attn_wmma(
    const _Float16* __restrict__ qh, const _Float16* __restrict__ kh,
    const _Float16* __restrict__ vtg, _Float16* __restrict__ yh) {
  __shared__ _Float16 kT[2][32 * HD];   // [key][d]  B-frag layout for scores
  __shared__ _Float16 vT[2][HD * 32];   // [d][key]  B-frag layout for P*V
  __shared__ _Float16 pT[128 * 32];     // probs staging [query][key]

  const int tid  = threadIdx.x;
  const int lane = tid & 31;
  const int wv   = tid >> 5;   // 8 waves, 16 queries each
  const int l15  = lane & 15;
  const int sub  = lane >> 4;

  const int h     = blockIdx.y;
  const int kvh   = h >> 2;  // GQA: H/HKV = 4
  const int qBase = blockIdx.x * 128;

  // ---- preload this wave's q A-fragments (16 queries x 128 d = 4 frags) ----
  v16h qa[4];
  {
    const int qrow = qBase + wv * 16 + l15;
#pragma unroll
    for (int dt = 0; dt < 4; ++dt) {
      const _Float16* p = qh + ((size_t)qrow * NH + h) * HD + dt * 32 + sub * 8;
      qa[dt] = cat8(*(const v8h*)p, *(const v8h*)(p + 16));
    }
  }

  v8f out[8];
#pragma unroll
  for (int dt = 0; dt < 8; ++dt) out[dt] = zero8f();
  float rowMax[8], rowSum[8];
#pragma unroll
  for (int i = 0; i < 8; ++i) { rowMax[i] = -1e30f; rowSum[i] = 0.0f; }

  int kfirst = qBase - (WIN - 1);
  if (kfirst < 0) kfirst = 0;
  const int kt0     = kfirst & ~31;
  const int ntiles  = (qBase + 128 - kt0 + 31) >> 5;
  const float scale = 0.08838834764831845f;  // 1/sqrt(128)

  const unsigned ldsK0 = (unsigned)(size_t)(void*)&kT[0][0];
  const unsigned ldsK1 = (unsigned)(size_t)(void*)&kT[1][0];
  const unsigned ldsV0 = (unsigned)(size_t)(void*)&vT[0][0];
  const unsigned ldsV1 = (unsigned)(size_t)(void*)&vT[1][0];

  // ---- prologue: DMA tile 0 into buffer 0 ----
  if (wv == 0) attn_issue_tiles(kh, vtg, kvh, kt0, ldsK0, ldsV0);

  for (int it = 0; it < ntiles; ++it) {
    const int cur = it & 1;
    const int keyBase = kt0 + it * 32;

    if (wv == 0) {
      if (it + 1 < ntiles) {
        attn_issue_tiles(kh, vtg, kvh, keyBase + 32,
                         cur ? ldsK0 : ldsK1, cur ? ldsV0 : ldsV1);
        __builtin_amdgcn_s_wait_tensorcnt(2);  // tile `it` complete (in-order)
      } else {
        __builtin_amdgcn_s_wait_tensorcnt(0);
      }
    }
    __syncthreads();

    // ---- scores: two 16-key n-tiles, K loop over D in 32-chunks ----
    v8f s0 = zero8f(), s1 = zero8f();
#pragma unroll
    for (int dt = 0; dt < 4; ++dt) {
      const int koff = dt * 32 + sub * 8;
      const _Float16* kp0 = &kT[cur][l15 * HD + koff];
      const _Float16* kp1 = &kT[cur][(16 + l15) * HD + koff];
      v16h b0 = cat8(*(const v8h*)kp0, *(const v8h*)(kp0 + 16));
      v16h b1 = cat8(*(const v8h*)kp1, *(const v8h*)(kp1 + 16));
      s0 = wmma_f16(qa[dt], b0, s0);
      s1 = wmma_f16(qa[dt], b1, s1);
    }

    // ---- online softmax (row stats replicated in each 16-lane half) ----
#pragma unroll
    for (int i = 0; i < 8; ++i) {
      const int p   = qBase + wv * 16 + sub * 8 + i;
      const int ka0 = keyBase + l15;
      const int ka1 = keyBase + 16 + l15;
      const bool v0 = (ka0 <= p) && (p - ka0 < WIN);
      const bool v1 = (ka1 <= p) && (p - ka1 < WIN);
      float f0 = v0 ? s0[i] * scale : -1e30f;
      float f1 = v1 ? s1[i] * scale : -1e30f;
      float t = fmaxf(f0, f1);
      t = fmaxf(t, __shfl_xor(t, 1, 32));
      t = fmaxf(t, __shfl_xor(t, 2, 32));
      t = fmaxf(t, __shfl_xor(t, 4, 32));
      t = fmaxf(t, __shfl_xor(t, 8, 32));
      const float newM  = fmaxf(rowMax[i], t);
      const float alpha = __expf(rowMax[i] - newM);
      const float p0 = v0 ? __expf(f0 - newM) : 0.0f;
      const float p1 = v1 ? __expf(f1 - newM) : 0.0f;
      float sum = p0 + p1;
      sum += __shfl_xor(sum, 1, 32);
      sum += __shfl_xor(sum, 2, 32);
      sum += __shfl_xor(sum, 4, 32);
      sum += __shfl_xor(sum, 8, 32);
      rowSum[i] = rowSum[i] * alpha + sum;
      rowMax[i] = newM;
#pragma unroll
      for (int dt = 0; dt < 8; ++dt) out[dt][i] *= alpha;
      const int prow = wv * 16 + sub * 8 + i;
      pT[prow * 32 + l15]      = (_Float16)p0;
      pT[prow * 32 + 16 + l15] = (_Float16)p1;
    }
    // pT rows are wave-private; same-wave LDS ordering makes them visible.

    // ---- out += P (16x32) * V (32x16 per d-tile) ----
    {
      const _Float16* pp = &pT[(wv * 16 + l15) * 32 + sub * 8];
      v16h pa = cat8(*(const v8h*)pp, *(const v8h*)(pp + 16));
#pragma unroll
      for (int dt = 0; dt < 8; ++dt) {
        const _Float16* vp = &vT[cur][(dt * 16 + l15) * 32 + sub * 8];
        v16h vb = cat8(*(const v8h*)vp, *(const v8h*)(vp + 16));
        out[dt] = wmma_f16(pa, vb, out[dt]);
      }
    }
    __syncthreads();
  }

  // ---- normalize and write y[(t, h, d)] as f16 (A of the Wo GEMM) ----
#pragma unroll
  for (int i = 0; i < 8; ++i) {
    const float inv = 1.0f / rowSum[i];
    const int p = qBase + wv * 16 + sub * 8 + i;
    _Float16* dst = yh + ((size_t)p * NH + h) * HD;
#pragma unroll
    for (int dt = 0; dt < 8; ++dt)
      dst[dt * 16 + l15] = (_Float16)(out[dt][i] * inv);
  }
}

// ======================================================================
extern "C" void kernel_launch(void* const* d_in, const int* in_sizes, int n_in,
                              void* d_out, int out_size, void* d_ws, size_t ws_size,
                              hipStream_t stream) {
  const float* x    = (const float*)d_in[0];
  const float* cosb = (const float*)d_in[1];
  const float* sinb = (const float*)d_in[2];
  const float* Wq   = (const float*)d_in[3];
  const float* Wk   = (const float*)d_in[4];
  const float* Wv   = (const float*)d_in[5];
  const float* Wo   = (const float*)d_in[6];
  float* out        = (float*)d_out;

  char* ws = (char*)d_ws;
  float* qf = (float*)ws;         ws += (size_t)T_LEN * NEMB * sizeof(float);
  float* kf = (float*)ws;         ws += (size_t)T_LEN * NKV * HD * sizeof(float);
  float* vf = (float*)ws;         ws += (size_t)T_LEN * NKV * HD * sizeof(float);
  _Float16* yh  = (_Float16*)ws;  ws += (size_t)T_LEN * NEMB * sizeof(_Float16);
  _Float16* qh  = (_Float16*)ws;  ws += (size_t)T_LEN * NEMB * sizeof(_Float16);
  _Float16* kh  = (_Float16*)ws;  ws += (size_t)T_LEN * NKV * HD * sizeof(_Float16);
  _Float16* vtg = (_Float16*)ws;  ws += (size_t)T_LEN * NKV * HD * sizeof(_Float16);

  const dim3 blk256(256);
  // QKV projections (fp32 A)
  gemm_wmma<NEMB, NEMB><<<dim3(NEMB / 128, T_LEN / 256), blk256, 0, stream>>>(
      x, Wq, qf);
  gemm_wmma<NKV * HD, NEMB><<<dim3((NKV * HD) / 128, T_LEN / 256), blk256, 0, stream>>>(
      x, Wk, kf);
  gemm_wmma<NKV * HD, NEMB><<<dim3((NKV * HD) / 128, T_LEN / 256), blk256, 0, stream>>>(
      x, Wv, vf);
  // RoPE / RMSNorm / quant-dequant -> f16 (v transposed to (HKV, D, T))
  rope_norm_quant<<<dim3(T_LEN, NH), 128, 0, stream>>>(qf, cosb, sinb, qh, NH, 0);
  rope_norm_quant<<<dim3(T_LEN, NKV), 128, 0, stream>>>(kf, cosb, sinb, kh, NKV, 1);
  rope_norm_quant<<<dim3(T_LEN, NKV), 128, 0, stream>>>(vf, cosb, sinb, vtg, NKV, 2);
  // Sliding-window attention (TDM double-buffered K/V) -> f16 y
  swa_attn_wmma<<<dim3(T_LEN / 128, NH), blk256, 0, stream>>>(qh, kh, vtg, yh);
  // Output projection (f16 A path: no reconversion, half the A traffic)
  gemm_wmma<NEMB, NEMB><<<dim3(NEMB / 128, T_LEN / 256), blk256, 0, stream>>>(
      yh, Wo, out);
}
